// InputRotationWrapper_90692529423063
// MI455X (gfx1250) — matrix operand order repaired and checked by reading, not compiled, not run on hardware
//
#include <hip/hip_runtime.h>

// ---------------------------------------------------------------------------
// InputRotationWrapper: out = (Hadamard(x)/64) @ W^T + b
//   x: (8192, 4096) f32   W: (4096,4096) f32   b: (4096,) f32  -> out f32
// WHT+downconvert to bf16, W->bf16, WMMA bf16 GEMM (f32 accum).
// GEMM: 128x256 block tile, 8 waves of 64x64 -> 16 v_wmma per 32-wide K-step,
// double-buffered LDS fed by GLOBAL_LOAD_ASYNC_TO_LDS_B128 when available.
// ---------------------------------------------------------------------------

typedef __attribute__((ext_vector_type(16))) __bf16 v16bf;
typedef __attribute__((ext_vector_type(8)))  float  v8f;

#define D_DIM 4096
#define BM 128
#define BN 256
#define BK 32
#define LDK 40   // padded LDS K-stride in bf16 elems (80B rows, keeps 16B align)

#if defined(__has_builtin)
#if __has_builtin(__builtin_amdgcn_global_load_async_to_lds_b128) && \
    __has_builtin(__builtin_amdgcn_s_wait_asynccnt)
#define USE_ASYNC_LDS 1
#endif
#endif
#ifndef USE_ASYNC_LDS
#define USE_ASYNC_LDS 0
#endif

#if USE_ASYNC_LDS
typedef __attribute__((vector_size(16))) int v4i_vs;
typedef __attribute__((address_space(1))) v4i_vs as1_v4i;  // global
typedef __attribute__((address_space(3))) v4i_vs as3_v4i;  // LDS

__device__ __forceinline__ void async_cp16(const void* g, const void* l) {
  // generic->AS(1)/AS(3) via integer casts: low 32 bits of a generic LDS
  // pointer are the LDS byte offset (flat aperture truncation rule).
  __builtin_amdgcn_global_load_async_to_lds_b128(
      (as1_v4i*)(uintptr_t)g,
      (as3_v4i*)(unsigned)(uintptr_t)l,
      0, 0);
}
#endif

__device__ __forceinline__ void wht16(float v[16]) {
#pragma unroll
  for (int h = 1; h < 16; h <<= 1) {
#pragma unroll
    for (int i = 0; i < 16; ++i) {
      if ((i & h) == 0) {
        float a = v[i], b = v[i + h];
        v[i]     = a + b;
        v[i + h] = a - b;
      }
    }
  }
}

// One block per row. 4096 = 16*16*16 -> three radix-16 register passes.
__global__ __launch_bounds__(256) void had_bf16_kernel(const float* __restrict__ x,
                                                       __bf16* __restrict__ out) {
  __shared__ float lds[D_DIM];
  const int t = threadIdx.x;
  const size_t row = blockIdx.x;
  const float* xr = x + row * (size_t)D_DIM;
  float v[16];

  {  // pass 1: low digit (contiguous 16)
    const float4* p = (const float4*)(xr + t * 16);
#pragma unroll
    for (int i = 0; i < 4; ++i) {
      float4 f = p[i];
      v[4 * i + 0] = f.x; v[4 * i + 1] = f.y;
      v[4 * i + 2] = f.z; v[4 * i + 3] = f.w;
    }
    wht16(v);
#pragma unroll
    for (int j = 0; j < 16; ++j) lds[t * 16 + j] = v[j];
  }
  __syncthreads();

  {  // pass 2: middle digit; i = a*256 + b*16 + c, thread owns (a, c)
    const int c = t & 15, a = t >> 4;
#pragma unroll
    for (int b_ = 0; b_ < 16; ++b_) v[b_] = lds[a * 256 + b_ * 16 + c];
    wht16(v);
#pragma unroll
    for (int b_ = 0; b_ < 16; ++b_) lds[a * 256 + b_ * 16 + c] = v[b_];
  }
  __syncthreads();

  {  // pass 3: high digit; thread owns low-8 index t; fuse 1/64 scale
#pragma unroll
    for (int a = 0; a < 16; ++a) v[a] = lds[a * 256 + t];
    wht16(v);
#pragma unroll
    for (int a = 0; a < 16; ++a) lds[a * 256 + t] = v[a] * 0.015625f;
  }
  __syncthreads();

  {  // coalesced bf16 write-out: 32B per lane
    alignas(16) __bf16 ob[16];
#pragma unroll
    for (int j = 0; j < 16; ++j) ob[j] = (__bf16)lds[t * 16 + j];
    uint4* dst = (uint4*)(out + row * (size_t)D_DIM + t * 16);
    const uint4* src = (const uint4*)ob;
    dst[0] = src[0];
    dst[1] = src[1];
  }
}

__global__ __launch_bounds__(256) void cvt_bf16_kernel(const float* __restrict__ w,
                                                       __bf16* __restrict__ o, int n4) {
  int i = blockIdx.x * blockDim.x + threadIdx.x;
  if (i < n4) {
    float4 f = ((const float4*)w)[i];
    alignas(8) __bf16 b4[4] = {(__bf16)f.x, (__bf16)f.y, (__bf16)f.z, (__bf16)f.w};
    ((uint2*)o)[i] = *(const uint2*)b4;
  }
}

// C = A(MxK) * B(NxK)^T + bias, bf16 in / f32 out.
// 256 threads = 8 waves (wave32); wave grid 2(M)x4(N); wave tile 64x64
// -> 4x4 = 16 x v_wmma_f32_16x16x32_bf16 per 32-wide K-step.
__global__ __launch_bounds__(256) void gemm_bf16_kernel(
    const __bf16* __restrict__ A, const __bf16* __restrict__ B,
    const float* __restrict__ bias, float* __restrict__ out) {
  __shared__ alignas(16) __bf16 As[2][BM][LDK];
  __shared__ alignas(16) __bf16 Bs[2][BN][LDK];

  const int t      = threadIdx.x;
  const int lane   = t & 31;
  const int lane16 = lane & 15;
  const int khalf  = lane >> 4;          // which K-half this lane holds
  const int wave   = t >> 5;
  const int wm     = wave & 1;           // *64 in M
  const int wn     = wave >> 1;          // *64 in N
  const int    blockN = blockIdx.x * BN;
  const size_t blockM = (size_t)blockIdx.y * BM;

  // A tile: 128 rows x 64B -> thread covers 32B of row t/2
  // B tile: 256 rows x 64B -> thread covers all 64B of row t
  const int arow  = t >> 1;
  const int ahalf = t & 1;
  const uint4* agp = (const uint4*)(A + (blockM + arow) * (size_t)D_DIM) + ahalf * 2;
  const uint4* bgp = (const uint4*)(B + (size_t)(blockN + t) * D_DIM);

  v8f acc[4][4];
#pragma unroll
  for (int mt = 0; mt < 4; ++mt)
#pragma unroll
    for (int nt = 0; nt < 4; ++nt) acc[mt][nt] = {};

  auto compute = [&](int buf) {
    union Fr { uint4 q[2]; v16bf v; };
    Fr fa[4], fb[4];
    // A frag (16x32): lane<16: M=lane, K in {0..7,16..23}; lane>=16: M=lane-16,
    // K in {8..15,24..31}  -> two b128 LDS loads per tile
#pragma unroll
    for (int mt = 0; mt < 4; ++mt) {
      const __bf16* p = &As[buf][wm * 64 + mt * 16 + lane16][khalf * 8];
      fa[mt].q[0] = *(const uint4*)(p);
      fa[mt].q[1] = *(const uint4*)(p + 16);
    }
    // B frag (32x16): N=lane16, K in {0..7,16..23} (lane<16) or {8..15,24..31}
#pragma unroll
    for (int nt = 0; nt < 4; ++nt) {
      const __bf16* p = &Bs[buf][wn * 64 + nt * 16 + lane16][khalf * 8];
      fb[nt].q[0] = *(const uint4*)(p);
      fb[nt].q[1] = *(const uint4*)(p + 16);
    }
#pragma unroll
    for (int mt = 0; mt < 4; ++mt)
#pragma unroll
      for (int nt = 0; nt < 4; ++nt)
        acc[mt][nt] = __builtin_amdgcn_wmma_f32_16x16x32_bf16(
            false, fa[mt].v, false, fb[nt].v, (short)0, acc[mt][nt], false, false);
  };

  const int nsteps = D_DIM / BK;  // 128

#if USE_ASYNC_LDS
  auto issueTile = [&](int buf, int kt) {
    const uint4* ap = agp + (size_t)kt * 4;  // BK bf16 = 64B = 4 uint4
    uint4* as = (uint4*)(&As[buf][arow][ahalf * 16]);
    async_cp16(ap + 0, as + 0);
    async_cp16(ap + 1, as + 1);
    const uint4* bp = bgp + (size_t)kt * 4;
    uint4* bs = (uint4*)(&Bs[buf][t][0]);
    async_cp16(bp + 0, bs + 0);
    async_cp16(bp + 1, bs + 1);
    async_cp16(bp + 2, bs + 2);
    async_cp16(bp + 3, bs + 3);
  };

  issueTile(0, 0);
#pragma unroll 1
  for (int kt = 0; kt < nsteps; ++kt) {
    const int cur = kt & 1;
    __builtin_amdgcn_s_wait_asynccnt(0);  // batch kt landed in LDS (own wave)
    __syncthreads();                      // all waves landed; prev reads done
    if (kt + 1 < nsteps) issueTile(cur ^ 1, kt + 1);  // overlap with WMMAs
    compute(cur);
  }
#else
  uint4 ra0, ra1, rb0, rb1, rb2, rb3;
  auto loadTile = [&](int kt) {
    const uint4* ap = agp + (size_t)kt * 4;
    ra0 = ap[0]; ra1 = ap[1];
    const uint4* bp = bgp + (size_t)kt * 4;
    rb0 = bp[0]; rb1 = bp[1]; rb2 = bp[2]; rb3 = bp[3];
  };
  auto storeTile = [&](int buf) {
    uint4* as = (uint4*)(&As[buf][arow][ahalf * 16]);
    as[0] = ra0; as[1] = ra1;
    uint4* bs = (uint4*)(&Bs[buf][t][0]);
    bs[0] = rb0; bs[1] = rb1; bs[2] = rb2; bs[3] = rb3;
  };

  loadTile(0);
  storeTile(0);
  __syncthreads();
#pragma unroll 1
  for (int kt = 0; kt < nsteps; ++kt) {
    const int cur = kt & 1, nxt = cur ^ 1;
    if (kt + 1 < nsteps) loadTile(kt + 1);  // overlap global fetch with WMMA
    compute(cur);
    if (kt + 1 < nsteps) {
      storeTile(nxt);
      __syncthreads();
    }
  }
#endif

  // epilogue: C/D layout -> VGPR r: M = r (lanes<16) or r+8 ; N = lane16
#pragma unroll
  for (int nt = 0; nt < 4; ++nt) {
    const int col = blockN + wn * 64 + nt * 16 + lane16;
    const float bv = bias[col];
#pragma unroll
    for (int mt = 0; mt < 4; ++mt) {
#pragma unroll
      for (int r = 0; r < 8; ++r) {
        const size_t rr = blockM + wm * 64 + mt * 16 + khalf * 8 + r;
        out[rr * D_DIM + col] = acc[mt][nt][r] + bv;
      }
    }
  }
}

extern "C" void kernel_launch(void* const* d_in, const int* in_sizes, int n_in,
                              void* d_out, int out_size, void* d_ws, size_t ws_size,
                              hipStream_t stream) {
  const float* x = (const float*)d_in[0];
  const float* W = (const float*)d_in[1];
  const float* b = (const float*)d_in[2];
  float* out = (float*)d_out;

  const int M = in_sizes[0] / D_DIM;  // 8192

  __bf16* Abf = (__bf16*)d_ws;  // 64 MB
  __bf16* Wbf = (__bf16*)((char*)d_ws + (size_t)M * D_DIM * sizeof(__bf16));  // +32 MB

  had_bf16_kernel<<<M, 256, 0, stream>>>(x, Abf);

  const int n4 = D_DIM * D_DIM / 4;
  cvt_bf16_kernel<<<(n4 + 255) / 256, 256, 0, stream>>>(W, Wbf, n4);

  dim3 grid(D_DIM / BN, M / BM);
  gemm_bf16_kernel<<<grid, 256, 0, stream>>>(Abf, Wbf, b, out);
}
